// Encoder_simple_18305150616328
// MI455X (gfx1250) — matrix operand order repaired and checked by reading, not compiled, not run on hardware
//
#include <hip/hip_runtime.h>

#define DIM 128
#define LSTR 136      // LDS row stride in bf16 elements (128 + 8 pad)
#define NLAYERS 3

typedef __bf16 bf16_t;
typedef __attribute__((ext_vector_type(16))) __bf16 v16bf;
typedef __attribute__((ext_vector_type(8)))  __bf16 v8bf;
typedef __attribute__((ext_vector_type(8)))  float  v8f;
typedef __attribute__((ext_vector_type(4))) unsigned int u32x4;
typedef __attribute__((ext_vector_type(8))) int         i32x8;
typedef __attribute__((ext_vector_type(4))) int         i32x4;

// ---------- helpers ----------
static __device__ __forceinline__ unsigned short f2bfu(float f) {
  unsigned int u = __builtin_bit_cast(unsigned int, f);
  u += 0x7FFFu + ((u >> 16) & 1u);          // round-to-nearest-even
  return (unsigned short)(u >> 16);
}
static __device__ __forceinline__ bf16_t f2bf(float f) {
  unsigned short h = f2bfu(f);
  return __builtin_bit_cast(bf16_t, h);
}

static __device__ __forceinline__ v8f wmma_bf16(v16bf a, v16bf b, v8f c) {
  // (neg_a, A, neg_b, B, c_mod, C, reuse_a, reuse_b)
  return __builtin_amdgcn_wmma_f32_16x16x32_bf16(false, a, false, b, (short)0, c,
                                                 false, false);
}

// TDM: DMA a contiguous run of `nelem2B` bf16/u16 elements (<= 65535) from
// global memory into contiguous LDS. D# per ISA ch.8:
//  group0: [1:0]count=1, [63:32]lds_addr, [120:64]global_addr, [127:126]type=2
//  group1: [17:16]data_size=1(2B), dim0/tile0 = nelem, dim1/tile1 = 1,
//          tensor_dim0_stride = nelem
// This toolchain exposes the 6-arg builtin:
//   (u32x4 g0, i32x8 g1, i32x4 g2, i32x4 g3, i32x8 g4, i32 cpol)
static __device__ __forceinline__ void tdm_load_to_lds(const void* gsrc,
                                                       void* lds_dst,
                                                       unsigned int nelem2B) {
  unsigned long long ga = (unsigned long long)(uintptr_t)gsrc;
  unsigned int lds = (unsigned int)(uintptr_t)lds_dst;  // low 32 = LDS offset
  u32x4 g0;
  g0[0] = 1u;                                            // count=1, user mode
  g0[1] = lds;
  g0[2] = (unsigned int)(ga & 0xFFFFFFFFu);
  g0[3] = (unsigned int)((ga >> 32) & 0x01FFFFFFu) | (2u << 30);  // type=2
  i32x8 g1;
  g1[0] = (int)(1u << 16);                               // data_size=2B, mask=0
  g1[1] = (int)((nelem2B & 0xFFFFu) << 16);              // tensor_dim0[15:0]
  g1[2] = (int)((nelem2B >> 16) | (1u << 16));           // dim0[31:16], dim1=1
  g1[3] = (int)((nelem2B & 0xFFFFu) << 16);              // tile_dim0
  g1[4] = 1;                                             // tile_dim1=1, tile2=0
  g1[5] = (int)nelem2B;                                  // dim0_stride[31:0]
  g1[6] = 0;
  g1[7] = 0;
  i32x4 g2 = {0, 0, 0, 0};
  i32x4 g3 = {0, 0, 0, 0};
  i32x8 g4 = {0, 0, 0, 0, 0, 0, 0, 0};
  __builtin_amdgcn_tensor_load_to_lds(g0, g1, g2, g3, g4, 0);
}

// A fragment: 16x32 bf16, row = lane&15, lane-half splits K per ISA layout:
// lanes 0-15: K 0-7 then 16-23 ; lanes 16-31: K 8-15 then 24-31 (+ kt*32)
static __device__ __forceinline__ v16bf load_a_frag(const bf16_t* tile, int lane,
                                                    int rowtile, int kt) {
  const int r  = (rowtile << 4) + (lane & 15);
  const int kb = (kt << 5) + ((lane >> 4) << 3);
  const bf16_t* p = tile + r * LSTR + kb;
  v8bf lo = *(const v8bf*)p;          // K kb..kb+7
  v8bf hi = *(const v8bf*)(p + 16);   // K kb+16..kb+23
  v16bf a;
#pragma unroll
  for (int i = 0; i < 8; ++i) { a[i] = lo[i]; a[i + 8] = hi[i]; }
  return a;
}

// B fragment: pre-swizzled so each lane's 16 bf16 are contiguous (32B load)
static __device__ __forceinline__ v16bf load_b_frag(const bf16_t* w, int lane,
                                                    int kt, int nt) {
  return *(const v16bf*)(w + (((((kt << 3) + nt) << 5) + lane) << 4));
}

// ---------- small kernels ----------
__global__ void zero_f32_kernel(float* __restrict__ p, int n) {
  int i = blockIdx.x * blockDim.x + threadIdx.x;
  if (i < n) p[i] = 0.0f;
}

// Convert fp32 128x128 weight (K-major rows) -> bf16 B-fragment layout.
// B-matrix lane layout (16x16x32 bf16): lane&15 = column N, element i ->
// K = (i>>3)*16 + (lane>>4)*8 + (i&7) (+ kt*32), matching the A fragment.
__global__ void swizzle_w_kernel(const float* __restrict__ src,
                                 unsigned short* __restrict__ dst) {
  int d = blockIdx.x * 256 + threadIdx.x;   // 0..16383
  int i    = d & 15;
  int lane = (d >> 4) & 31;
  int nt   = (d >> 9) & 7;
  int kt   = d >> 12;
  int K  = (kt << 5) + ((i >> 3) << 4) + ((lane >> 4) << 3) + (i & 7);
  int Nc = (nt << 4) + (lane & 15);
  dst[d] = f2bfu(src[K * DIM + Nc]);
}

// Edge aggregation: agg[dst[e]] += x[src[e]]  (32 lanes x float4 per edge)
__global__ void agg_kernel(const float* __restrict__ x, const int* __restrict__ src,
                           const int* __restrict__ dst, float* __restrict__ agg,
                           int E) {
  int t = blockIdx.x * blockDim.x + threadIdx.x;
  int e = t >> 5;
  if (e >= E) return;
  int lane = t & 31;
  const float4 v = *(const float4*)(x + (long)src[e] * DIM + lane * 4);
  float* a = agg + (long)dst[e] * DIM + lane * 4;
  atomicAdd(a + 0, v.x);
  atomicAdd(a + 1, v.y);
  atomicAdd(a + 2, v.z);
  atomicAdd(a + 3, v.w);
}

// BN finalize: stats = [sum(128), sumsq(128)] -> ss = [scale(128), shift(128)]
__global__ void bn_finalize_kernel(const float* __restrict__ stats,
                                   const float* __restrict__ gamma,
                                   const float* __restrict__ beta,
                                   float* __restrict__ ss, float inv_n) {
  int c = threadIdx.x;  // 128 threads
  float m   = stats[c] * inv_n;
  float var = stats[DIM + c] * inv_n - m * m;
  float sc  = gamma[c] * rsqrtf(var + 1e-5f);
  ss[c]       = sc;
  ss[DIM + c] = beta[c] - m * sc;
}

// out = z * scale + shift   (float4 per thread)
__global__ void bn_apply_kernel(const float* __restrict__ zin,
                                const float* __restrict__ ss,
                                float* __restrict__ out, int n4) {
  int i = blockIdx.x * blockDim.x + threadIdx.x;
  if (i >= n4) return;
  long b = (long)i * 4;
  int c = (int)(b & 127);
  float4 v = *(const float4*)(zin + b);
  float4 r;
  r.x = v.x * ss[c + 0] + ss[DIM + c + 0];
  r.y = v.y * ss[c + 1] + ss[DIM + c + 1];
  r.z = v.z * ss[c + 2] + ss[DIM + c + 2];
  r.w = v.w * ss[c + 3] + ss[DIM + c + 3];
  *(float4*)(out + b) = r;
}

// Per-node: w = sigmoid(x.sw_w + sw_b); noisy = (1-w)x ; slots[batch] += w*x
__global__ void summary_kernel(const float* __restrict__ x,
                               const float* __restrict__ sw_w,
                               const float* __restrict__ sw_b,
                               const int* __restrict__ batch,
                               float* __restrict__ noisy,
                               float* __restrict__ slots, int nrows) {
  int node = blockIdx.x * 8 + (threadIdx.x >> 5);
  if (node >= nrows) return;
  int lane = threadIdx.x & 31;
  const float4 xv = *(const float4*)(x + (long)node * DIM + lane * 4);
  const float4 wv = *(const float4*)(sw_w + lane * 4);
  float d = xv.x * wv.x + xv.y * wv.y + xv.z * wv.z + xv.w * wv.w;
#pragma unroll
  for (int off = 16; off; off >>= 1) d += __shfl_xor(d, off, 32);
  float w = 1.0f / (1.0f + __expf(-(d + sw_b[0])));
  float4 s  = make_float4(w * xv.x, w * xv.y, w * xv.z, w * xv.w);
  float4 nv = make_float4(xv.x - s.x, xv.y - s.y, xv.z - s.z, xv.w - s.w);
  *(float4*)(noisy + (long)node * DIM + lane * 4) = nv;
  float* sp = slots + (long)batch[node] * DIM + lane * 4;
  atomicAdd(sp + 0, s.x);
  atomicAdd(sp + 1, s.y);
  atomicAdd(sp + 2, s.z);
  atomicAdd(sp + 3, s.w);
}

// ---------- fused WMMA GEMM (+GEMM2 when FUSED) + relu + BN-stats ----------
// FUSED=true : z = relu(relu((x+agg)@W1 + b1)@W2 + b2)   (GIN MLP body)
// FUSED=false: z = relu(x@W1 + b1)                        (VAE heads)
// Weights are streamed into LDS by the Tensor Data Mover (wave 0 issues the
// DMA, TENSORcnt-tracked) while all waves convert the input tile to bf16.
// Each block: 128 rows; each wave owns one 16-row stripe (fully private ->
// no barrier needed between GEMM1 epilogue and GEMM2 A reload).
template <bool FUSED>
__global__ __launch_bounds__(256, 2) void gemm_bn_kernel(
    const float* __restrict__ xin, const float* __restrict__ agg,
    const unsigned short* __restrict__ w1g, const float* __restrict__ b1,
    const unsigned short* __restrict__ w2g, const float* __restrict__ b2,
    float* __restrict__ zout, float* __restrict__ stats, int nrows) {
  __shared__ __align__(16) bf16_t W1[DIM * DIM];
  __shared__ __align__(16) bf16_t W2[FUSED ? DIM * DIM : 16];
  __shared__ __align__(16) bf16_t tile[128 * LSTR];
  __shared__ float red[2 * DIM];

  const int tid  = threadIdx.x;
  const int lane = tid & 31;
  const int wv   = tid >> 5;
  const long row0 = (long)blockIdx.x * 128;

  // Kick off TDM weight DMA from wave 0 (EXEC-ignored wave-level op).
  if (wv == 0) {
    tdm_load_to_lds(w1g, W1, DIM * DIM);
    if (FUSED) tdm_load_to_lds(w2g, W2, DIM * DIM);
  }
  if (tid < 2 * DIM) red[tid] = 0.0f;

  {  // load input tile as bf16; clamp tail rows (stores predicated later)
    const int r  = tid >> 1;
    const int c0 = (tid & 1) << 6;
    long grow = row0 + r;
    if (grow >= nrows) grow = nrows - 1;
    long prow = row0 + 128 + r;           // prefetch next row-block
    if (prow >= nrows) prow = nrows - 1;
    __builtin_prefetch(xin + prow * DIM + c0, 0, 3);
    const float4* xp = (const float4*)(xin + grow * DIM + c0);
    bf16_t* tp = tile + r * LSTR + c0;
    if (FUSED) {
      __builtin_prefetch(agg + prow * DIM + c0, 0, 3);
      const float4* ap = (const float4*)(agg + grow * DIM + c0);
#pragma unroll
      for (int j = 0; j < 16; ++j) {
        float4 xv = xp[j], av = ap[j];
        tp[j * 4 + 0] = f2bf(xv.x + av.x);
        tp[j * 4 + 1] = f2bf(xv.y + av.y);
        tp[j * 4 + 2] = f2bf(xv.z + av.z);
        tp[j * 4 + 3] = f2bf(xv.w + av.w);
      }
    } else {
#pragma unroll
      for (int j = 0; j < 16; ++j) {
        float4 xv = xp[j];
        tp[j * 4 + 0] = f2bf(xv.x);
        tp[j * 4 + 1] = f2bf(xv.y);
        tp[j * 4 + 2] = f2bf(xv.z);
        tp[j * 4 + 3] = f2bf(xv.w);
      }
    }
  }
  // Issuing wave drains its TENSORcnt before joining the barrier; the
  // barrier then publishes the DMA'd weights to all waves.
  if (wv == 0) __builtin_amdgcn_s_wait_tensorcnt(0);
  __syncthreads();

  const int cloc = lane & 15;
  const int rloc = (wv << 4) + ((lane >> 4) << 3);

  v16bf a0 = load_a_frag(tile, lane, wv, 0);
  v16bf a1 = load_a_frag(tile, lane, wv, 1);
  v16bf a2 = load_a_frag(tile, lane, wv, 2);
  v16bf a3 = load_a_frag(tile, lane, wv, 3);

  if (FUSED) {
    // GEMM1 -> relu(+b1) -> bf16 back into the wave's own 16 tile rows
#pragma unroll
    for (int nt = 0; nt < 8; ++nt) {
      v8f c = {};
      c = wmma_bf16(a0, load_b_frag(W1, lane, 0, nt), c);
      c = wmma_bf16(a1, load_b_frag(W1, lane, 1, nt), c);
      c = wmma_bf16(a2, load_b_frag(W1, lane, 2, nt), c);
      c = wmma_bf16(a3, load_b_frag(W1, lane, 3, nt), c);
      const int col = (nt << 4) + cloc;
      const float bias = b1[col];
#pragma unroll
      for (int j = 0; j < 8; ++j) {
        float v = c[j] + bias;
        v = v > 0.0f ? v : 0.0f;
        tile[(rloc + j) * LSTR + col] = f2bf(v);
      }
    }
    // reload A fragments from freshly written t (same-wave LDS RAW)
    a0 = load_a_frag(tile, lane, wv, 0);
    a1 = load_a_frag(tile, lane, wv, 1);
    a2 = load_a_frag(tile, lane, wv, 2);
    a3 = load_a_frag(tile, lane, wv, 3);
  }

  const bf16_t* Wo = FUSED ? W2 : W1;
  const float*  bo = FUSED ? b2 : b1;
#pragma unroll
  for (int nt = 0; nt < 8; ++nt) {
    v8f c = {};
    c = wmma_bf16(a0, load_b_frag(Wo, lane, 0, nt), c);
    c = wmma_bf16(a1, load_b_frag(Wo, lane, 1, nt), c);
    c = wmma_bf16(a2, load_b_frag(Wo, lane, 2, nt), c);
    c = wmma_bf16(a3, load_b_frag(Wo, lane, 3, nt), c);
    const int col = (nt << 4) + cloc;
    const float bias = bo[col];
    float ls = 0.0f, lq = 0.0f;
#pragma unroll
    for (int j = 0; j < 8; ++j) {
      long grow = row0 + rloc + j;
      float v = c[j] + bias;
      v = v > 0.0f ? v : 0.0f;
      if (grow < nrows) {
        zout[grow * DIM + col] = v;
        ls += v;
        lq += v * v;
      }
    }
    atomicAdd(&red[col], ls);
    atomicAdd(&red[DIM + col], lq);
  }
  __syncthreads();
  if (tid < 2 * DIM) atomicAdd(&stats[tid], red[tid]);
}

// ---------- host ----------
extern "C" void kernel_launch(void* const* d_in, const int* in_sizes, int n_in,
                              void* d_out, int out_size, void* d_ws, size_t ws_size,
                              hipStream_t stream) {
  const float* x0      = (const float*)d_in[0];
  const int*   src     = (const int*)d_in[1];
  const int*   dst     = (const int*)d_in[2];
  const int*   batch   = (const int*)d_in[3];
  const float* conv_w1 = (const float*)d_in[5];
  const float* conv_b1 = (const float*)d_in[6];
  const float* conv_w2 = (const float*)d_in[7];
  const float* conv_b2 = (const float*)d_in[8];
  const float* bng     = (const float*)d_in[9];
  const float* bnb     = (const float*)d_in[10];
  const float* sw_w    = (const float*)d_in[11];
  const float* sw_b    = (const float*)d_in[12];
  const float* nmu_w   = (const float*)d_in[13];
  const float* nmu_b   = (const float*)d_in[14];
  const float* nlv_w   = (const float*)d_in[15];
  const float* nlv_b   = (const float*)d_in[16];
  const float* gmu_w   = (const float*)d_in[17];
  const float* gmu_b   = (const float*)d_in[18];
  const float* glv_w   = (const float*)d_in[19];
  const float* glv_b   = (const float*)d_in[20];
  const float* hgam    = (const float*)d_in[21];
  const float* hbet    = (const float*)d_in[22];
  float* out = (float*)d_out;

  const int N = in_sizes[0] / DIM;
  const int E = in_sizes[1];
  const int G = (out_size - 2 * N * DIM) / (2 * DIM);

  // workspace carve-up (256B aligned)
  char* ws = (char*)d_ws;
  size_t o = 0;
  auto carve = [&](size_t bytes) {
    char* p = ws + o;
    o = (o + bytes + 255) & ~(size_t)255;
    return p;
  };
  float* xbuf   = (float*)carve((size_t)N * DIM * 4);
  float* aggbuf = (float*)carve((size_t)N * DIM * 4);  // reused as noisy_info
  float* zbuf   = (float*)carve((size_t)N * DIM * 4);
  float* slots  = (float*)carve((size_t)G * DIM * 4);
  float* stats  = (float*)carve(256 * 4);
  float* ssbuf  = (float*)carve(256 * 4);
  unsigned short* wbf = (unsigned short*)carve((size_t)10 * DIM * DIM * 2);

  // 0) convert + swizzle all weight matrices to bf16 B-fragment layout
  const float* wsrc[10] = {conv_w1,                conv_w1 + DIM * DIM,
                           conv_w1 + 2 * DIM * DIM, conv_w2,
                           conv_w2 + DIM * DIM,     conv_w2 + 2 * DIM * DIM,
                           nmu_w, nlv_w, gmu_w, glv_w};
  for (int m = 0; m < 10; ++m)
    swizzle_w_kernel<<<64, 256, 0, stream>>>(wsrc[m], wbf + (size_t)m * DIM * DIM);

  const int nElem  = N * DIM;
  const int gElem  = G * DIM;
  const int gemmBlocksN = (N + 127) / 128;
  const int gemmBlocksG = (G + 127) / 128;

  // 1) GIN layers
  const float* xcur = x0;
  for (int l = 0; l < NLAYERS; ++l) {
    zero_f32_kernel<<<(nElem + 255) / 256, 256, 0, stream>>>(aggbuf, nElem);
    agg_kernel<<<(E * 32 + 255) / 256, 256, 0, stream>>>(xcur, src, dst, aggbuf, E);
    zero_f32_kernel<<<1, 256, 0, stream>>>(stats, 256);
    gemm_bn_kernel<true><<<gemmBlocksN, 256, 0, stream>>>(
        xcur, aggbuf, wbf + (size_t)l * DIM * DIM, conv_b1 + l * DIM,
        wbf + (size_t)(3 + l) * DIM * DIM, conv_b2 + l * DIM, zbuf, stats, N);
    bn_finalize_kernel<<<1, 128, 0, stream>>>(stats, bng + l * DIM, bnb + l * DIM,
                                              ssbuf, 1.0f / (float)N);
    bn_apply_kernel<<<(nElem / 4 + 255) / 256, 256, 0, stream>>>(zbuf, ssbuf, xbuf,
                                                                 nElem / 4);
    xcur = xbuf;
  }

  // 2) summary weighter: noisy (in aggbuf) + slot scatter
  zero_f32_kernel<<<(gElem + 255) / 256, 256, 0, stream>>>(slots, gElem);
  summary_kernel<<<(N + 7) / 8, 256, 0, stream>>>(xbuf, sw_w, sw_b, batch, aggbuf,
                                                  slots, N);

  // 3) four heads: relu(in@W+b) -> BN -> concatenated output
  struct Head {
    const float* in; int rows; int wmat; const float* b;
    const float* g; const float* be; float* o;
  };
  Head heads[4] = {
      {aggbuf, N, 6, nmu_b, hgam + 0 * DIM, hbet + 0 * DIM, out},
      {aggbuf, N, 7, nlv_b, hgam + 1 * DIM, hbet + 1 * DIM, out + (size_t)N * DIM},
      {slots,  G, 8, gmu_b, hgam + 2 * DIM, hbet + 2 * DIM, out + (size_t)2 * N * DIM},
      {slots,  G, 9, glv_b, hgam + 3 * DIM, hbet + 3 * DIM,
       out + (size_t)2 * N * DIM + (size_t)G * DIM},
  };
  for (int h = 0; h < 4; ++h) {
    const Head& H = heads[h];
    const int blocks = (H.rows == N) ? gemmBlocksN : gemmBlocksG;
    const int elems  = H.rows * DIM;
    zero_f32_kernel<<<1, 256, 0, stream>>>(stats, 256);
    gemm_bn_kernel<false><<<blocks, 256, 0, stream>>>(
        H.in, nullptr, wbf + (size_t)H.wmat * DIM * DIM, H.b, nullptr, nullptr,
        zbuf, stats, H.rows);
    bn_finalize_kernel<<<1, 128, 0, stream>>>(stats, H.g, H.be, ssbuf,
                                              1.0f / (float)H.rows);
    bn_apply_kernel<<<(elems / 4 + 255) / 256, 256, 0, stream>>>(zbuf, ssbuf, H.o,
                                                                 elems / 4);
  }
  (void)n_in; (void)ws_size;
}
